// GRU_73100343378118
// MI455X (gfx1250) — compile-verified, matrix operand order checked by reading, not run on hardware
//
#include <hip/hip_runtime.h>
#include <cstdint>
#include <cstddef>

#define INPUT   512
#define HIDDEN  512
#define G3      1536      // 3*HIDDEN
#define ROLLOUT 32
#define NTRAJ   1024

#define SA   40           // LDS stride (bf16) for staged A tile in GEMM (128x32, padded)
#define BBLK 16           // trajectories per workgroup in recurrence
#define SHB  520          // LDS stride (bf16) for h bf16 copy (512 + 8 pad)
#define SGH  1540         // LDS stride (f32) for gh scratch (1536 + 4 pad)

typedef __attribute__((ext_vector_type(16))) __bf16 v16bf;
typedef __attribute__((ext_vector_type(8)))  __bf16 bf16x8;
typedef __attribute__((ext_vector_type(8)))  float  v8f;
typedef __attribute__((ext_vector_type(4)))  int    i32x4;

// ---------------------------------------------------------------------------
// CDNA5 async global->LDS copy (GLOBAL_LOAD_ASYNC_TO_LDS_B128, ASYNCcnt).
// Builtin signature (from clang diagnostic): param0 = AS(1) i32x4* (global src),
// param1 = AS(3) i32x4* (LDS dst), then two immediates (offset, cpol).
// Guarded: falls back to a synchronous bf16 copy if the builtin is absent.
// ---------------------------------------------------------------------------
#if defined(__gfx1250__) && __has_builtin(__builtin_amdgcn_global_load_async_to_lds_b128)
#define GRU_ASYNC_LDS 1
#endif

__device__ __forceinline__ void gl2lds_b128(const __bf16* gsrc, __bf16* ldst) {
#ifdef GRU_ASYNC_LDS
    __builtin_amdgcn_global_load_async_to_lds_b128(
        (__attribute__((address_space(1))) i32x4*)
            (__attribute__((address_space(1))) void*)(uintptr_t)(const void*)gsrc,
        (__attribute__((address_space(3))) i32x4*)
            (__attribute__((address_space(3))) void*)ldst,
        0, 0);
#else
    *(bf16x8*)ldst = *(const bf16x8*)gsrc;
#endif
}

__device__ __forceinline__ void wait_async0() {
#ifdef GRU_ASYNC_LDS
#if __has_builtin(__builtin_amdgcn_s_wait_asynccnt)
    __builtin_amdgcn_s_wait_asynccnt(0);
#else
    asm volatile("s_wait_asynccnt 0x0" ::: "memory");
#endif
#endif
}

// CDNA5 16-bit A/B fragment: lane = 16*khalf + idx holds K = khalf*8 + [0..7]
// and K = 16 + khalf*8 + [0..7] -> two contiguous 16B chunks, 16 elements apart.
__device__ __forceinline__ v16bf frag_load(const __bf16* p) {
    bf16x8 lo = *(const bf16x8*)(p);
    bf16x8 hi = *(const bf16x8*)(p + 16);
    v16bf r;
#pragma unroll
    for (int i = 0; i < 8; ++i) { r[i] = lo[i]; r[i + 8] = hi[i]; }
    return r;
}

__device__ __forceinline__ v8f v8f_zero() {
    v8f z;
#pragma unroll
    for (int i = 0; i < 8; ++i) z[i] = 0.0f;
    return z;
}

__device__ __forceinline__ float sigmoidf_(float x) {
    return 1.0f / (1.0f + __expf(-x));
}

// ---------------------------------------------------------------------------
// Kernel 0a: convert GRU weights fp32 -> bf16 into workspace (row-major [G3,K])
// ---------------------------------------------------------------------------
__global__ void GRU_wcvt_kernel(const float* __restrict__ wih,
                                const float* __restrict__ whh,
                                unsigned short* __restrict__ wih_bf_us,
                                unsigned short* __restrict__ whh_bf_us) {
    __bf16* wih_bf = (__bf16*)wih_bf_us;
    __bf16* whh_bf = (__bf16*)whh_bf_us;
    int i = blockIdx.x * blockDim.x + threadIdx.x;
    const int n = G3 * INPUT;
    if (i < n) {
        wih_bf[i] = (__bf16)wih[i];
        whh_bf[i] = (__bf16)whh[i];
    }
}

// ---------------------------------------------------------------------------
// Kernel 0b: convert head_output fp32 -> bf16 (8 elements / thread, vectorized)
// ---------------------------------------------------------------------------
__global__ void GRU_xcvt_kernel(const float* __restrict__ x,
                                unsigned short* __restrict__ xbf_us) {
    __bf16* xbf = (__bf16*)xbf_us;
    size_t i = ((size_t)blockIdx.x * blockDim.x + threadIdx.x) * 8;
    const float4 a = *(const float4*)(x + i);
    const float4 b = *(const float4*)(x + i + 4);
    bf16x8 o;
    o[0] = (__bf16)a.x; o[1] = (__bf16)a.y; o[2] = (__bf16)a.z; o[3] = (__bf16)a.w;
    o[4] = (__bf16)b.x; o[5] = (__bf16)b.y; o[6] = (__bf16)b.z; o[7] = (__bf16)b.w;
    *(bf16x8*)(xbf + i) = o;
}

// ---------------------------------------------------------------------------
// Kernel 1: gates_x = head_output @ W_ih^T + b_ih   (M=32768, N=1536, K=512)
// 128x128 tile per WG, 8 waves (2 along M x 4 along N), each wave 64x32.
// A (x bf16) staged via async global->LDS b128 copies, double-buffered;
// B (W_ih bf16) fragments straight from L2.
// ---------------------------------------------------------------------------
__global__ __launch_bounds__(256) void GRU_gxgemm_kernel(
        const unsigned short* __restrict__ xbf_us,
        const unsigned short* __restrict__ wih_bf_us,
        const float* __restrict__ b_ih,
        float* __restrict__ gx) {
    extern __shared__ char smem_raw[];
    __bf16* As = (__bf16*)smem_raw;                 // [2][128][SA]
    const __bf16* xbf    = (const __bf16*)xbf_us;
    const __bf16* wih_bf = (const __bf16*)wih_bf_us;

    const int tid    = threadIdx.x;
    const int lane   = tid & 31;
    const int wave   = tid >> 5;
    const int lane16 = lane & 15;
    const int khalf  = lane >> 4;
    const int m0 = blockIdx.x * 128;
    const int n0 = blockIdx.y * 128;
    const int wm = wave >> 2;                       // 0..1 -> 64 rows each
    const int wn = wave & 3;                        // 0..3 -> 32 cols each

    // Stage one 128x32 A tile (512 x 16B chunks; 2 per thread) into buffer `buf`.
    auto stage = [&](int buf, int k0) {
#pragma unroll
        for (int c = 0; c < 2; ++c) {
            int e  = c * 256 + tid;                 // chunk 0..511
            int r  = e >> 2;                        // row 0..127
            int kq = e & 3;                         // 16B chunk within 32-K slab
            const __bf16* g = xbf + (size_t)(m0 + r) * INPUT + k0 + kq * 8;
            __bf16* l = As + buf * (128 * SA) + r * SA + kq * 8;
            gl2lds_b128(g, l);
        }
    };

    v8f acc[4][2];
#pragma unroll
    for (int mt = 0; mt < 4; ++mt)
#pragma unroll
        for (int nt = 0; nt < 2; ++nt) acc[mt][nt] = v8f_zero();

    stage(0, 0);

    for (int kidx = 0; kidx < INPUT / 32; ++kidx) {
        const int k0  = kidx * 32;
        const int cur = kidx & 1;
        wait_async0();
        __syncthreads();                            // tile `cur` visible to all waves
        if (kidx + 1 < INPUT / 32)
            stage(cur ^ 1, k0 + 32);                // overlap next copy with compute

        const __bf16* Ab = As + cur * (128 * SA);
        v16bf afr[4], bfr[2];
#pragma unroll
        for (int mt = 0; mt < 4; ++mt)
            afr[mt] = frag_load(Ab + (wm * 64 + mt * 16 + lane16) * SA + khalf * 8);
#pragma unroll
        for (int nt = 0; nt < 2; ++nt)
            bfr[nt] = frag_load(wih_bf +
                (size_t)(n0 + wn * 32 + nt * 16 + lane16) * INPUT + k0 + khalf * 8);
#pragma unroll
        for (int mt = 0; mt < 4; ++mt)
#pragma unroll
            for (int nt = 0; nt < 2; ++nt)
                acc[mt][nt] = __builtin_amdgcn_wmma_f32_16x16x32_bf16(
                    false, afr[mt], false, bfr[nt], (short)0, acc[mt][nt], false, false);
    }

    // Epilogue: D layout -> M = khalf*8 + v, N = lane16 (per 16x16 tile)
#pragma unroll
    for (int mt = 0; mt < 4; ++mt)
#pragma unroll
        for (int nt = 0; nt < 2; ++nt) {
            int col = n0 + wn * 32 + nt * 16 + lane16;
            float bb = b_ih[col];
#pragma unroll
            for (int v = 0; v < 8; ++v) {
                int row = m0 + wm * 64 + mt * 16 + khalf * 8 + v;
                gx[(size_t)row * G3 + col] = acc[mt][nt][v] + bb;
            }
        }
}

// ---------------------------------------------------------------------------
// Kernel 2: masked GRU recurrence. 64 WGs x 256 threads; each WG owns 16
// trajectories and loops t=0..31 privately (no cross-WG dependency).
// LDS: h fp32 [16][512], h bf16 [16][SHB], gh fp32 [16][SGH]  (~145 KB, CDNA5
// 320 KB WGP LDS makes this possible).
// ---------------------------------------------------------------------------
__global__ __launch_bounds__(256) void GRU_rec_kernel(
        const float* __restrict__ h0,
        const float* __restrict__ dones,
        const unsigned short* __restrict__ whh_bf_us,
        const float* __restrict__ gx,
        const float* __restrict__ b_hh,
        float* __restrict__ out,
        float* __restrict__ h_last) {
    extern __shared__ char smem_raw[];
    float*  hS  = (float*)smem_raw;                                    // [16][512]
    __bf16* hB  = (__bf16*)(smem_raw + BBLK * 512 * 4);                // [16][SHB]
    float*  ghS = (float*)(smem_raw + BBLK * 512 * 4 + BBLK * SHB * 2);// [16][SGH]
    const __bf16* whh_bf = (const __bf16*)whh_bf_us;

    const int tid    = threadIdx.x;
    const int lane   = tid & 31;
    const int wave   = tid >> 5;                    // 0..7, each owns 192 N-cols
    const int lane16 = lane & 15;
    const int khalf  = lane >> 4;
    const int b0 = blockIdx.x * BBLK;

    // init h from rnn_states
#pragma unroll 8
    for (int it = 0; it < 32; ++it) {
        int e = it * 256 + tid;
        int b = e >> 9, j = e & 511;
        hS[b * 512 + j] = h0[(size_t)(b0 + b) * HIDDEN + j];
    }
    __syncthreads();

    for (int t = 0; t < ROLLOUT; ++t) {
        // episode-boundary mask (dones rolled by 1, first step = 0) + bf16 copy
#pragma unroll 8
        for (int it = 0; it < 32; ++it) {
            int e = it * 256 + tid;
            int b = e >> 9, j = e & 511;
            float m  = (t == 0) ? 0.0f : dones[(size_t)(b0 + b) * ROLLOUT + (t - 1)];
            float hv = hS[b * 512 + j] * (1.0f - m);
            hS[b * 512 + j] = hv;
            hB[b * SHB + j] = (__bf16)hv;
        }
        __syncthreads();

        // gh = h @ W_hh^T : each wave computes 16 x 192 (12 WMMA tiles), K=512
        v8f acc[12];
#pragma unroll
        for (int nt = 0; nt < 12; ++nt) acc[nt] = v8f_zero();

        for (int k0 = 0; k0 < HIDDEN; k0 += 32) {
            v16bf a = frag_load(hB + lane16 * SHB + k0 + khalf * 8);
#pragma unroll
            for (int nt = 0; nt < 12; ++nt) {
                v16bf bfrag = frag_load(whh_bf +
                    (size_t)(wave * 192 + nt * 16 + lane16) * HIDDEN + k0 + khalf * 8);
                acc[nt] = __builtin_amdgcn_wmma_f32_16x16x32_bf16(
                    false, a, false, bfrag, (short)0, acc[nt], false, false);
            }
        }
        // spill gh to LDS (stride padded so khalf halves hit disjoint banks)
#pragma unroll
        for (int nt = 0; nt < 12; ++nt) {
            int n = wave * 192 + nt * 16 + lane16;
#pragma unroll
            for (int v = 0; v < 8; ++v)
                ghS[(khalf * 8 + v) * SGH + n] = acc[nt][v];
        }
        __syncthreads();

        // gate math: r,z,n per (trajectory, hidden unit); update h; emit output
#pragma unroll 4
        for (int it = 0; it < 32; ++it) {
            int e = it * 256 + tid;
            int b = e >> 9, j = e & 511;
            size_t row = (size_t)(b0 + b) * ROLLOUT + t;
            const float* gxr = gx + row * G3;
            float ghr = ghS[b * SGH + j]        + b_hh[j];
            float ghz = ghS[b * SGH + 512 + j]  + b_hh[512 + j];
            float ghn = ghS[b * SGH + 1024 + j] + b_hh[1024 + j];
            float r  = sigmoidf_(gxr[j] + ghr);
            float z  = sigmoidf_(gxr[512 + j] + ghz);
            float hv = hS[b * 512 + j];
            float nn = tanhf(gxr[1024 + j] + r * ghn);
            float hn = (1.0f - z) * nn + z * hv;
            hS[b * 512 + j] = hn;
            out[row * HIDDEN + j] = hn;
        }
        __syncthreads();
    }

    // final hidden state
#pragma unroll 8
    for (int it = 0; it < 32; ++it) {
        int e = it * 256 + tid;
        int b = e >> 9, j = e & 511;
        h_last[(size_t)(b0 + b) * HIDDEN + j] = hS[b * 512 + j];
    }
}

// ---------------------------------------------------------------------------
extern "C" void kernel_launch(void* const* d_in, const int* in_sizes, int n_in,
                              void* d_out, int out_size, void* d_ws, size_t ws_size,
                              hipStream_t stream) {
    (void)in_sizes; (void)n_in; (void)out_size; (void)ws_size;
    const float* x    = (const float*)d_in[0];   // [NTRAJ*ROLLOUT, INPUT]
    const float* h0   = (const float*)d_in[1];   // [NTRAJ, HIDDEN]
    const float* dons = (const float*)d_in[2];   // [NTRAJ*ROLLOUT]
    const float* wih  = (const float*)d_in[3];   // [G3, INPUT]
    const float* whh  = (const float*)d_in[4];   // [G3, HIDDEN]
    const float* bih  = (const float*)d_in[5];   // [G3]
    const float* bhh  = (const float*)d_in[6];   // [G3]
    // d_in[7] = is_seq (1): sequence path implemented.

    float* out    = (float*)d_out;                                   // [32768,512]
    float* h_last = out + (size_t)NTRAJ * ROLLOUT * HIDDEN;          // [1024,512]

    char* ws = (char*)d_ws;
    unsigned short* wih_bf = (unsigned short*)ws;                    // 1.5 MB
    unsigned short* whh_bf = wih_bf + (size_t)G3 * INPUT;            // 1.5 MB
    unsigned short* xbf    = whh_bf + (size_t)G3 * INPUT;            // 32 MB
    float* gx = (float*)(ws + (size_t)2 * G3 * INPUT * 2
                            + (size_t)NTRAJ * ROLLOUT * INPUT * 2);  // 192 MB fp32

    // 0a) weights fp32 -> bf16
    GRU_wcvt_kernel<<<dim3((G3 * INPUT + 255) / 256), 256, 0, stream>>>(
        wih, whh, wih_bf, whh_bf);
    // 0b) x fp32 -> bf16 (8 elems/thread)
    GRU_xcvt_kernel<<<dim3((NTRAJ * ROLLOUT * INPUT / 8) / 256), 256, 0, stream>>>(
        x, xbf);

    // 1) gates_x GEMM: grid 256 (M tiles) x 12 (N tiles), double-buffered LDS
    GRU_gxgemm_kernel<<<dim3(256, 12), 256, 2 * 128 * SA * 2, stream>>>(
        xbf, wih_bf, bih, gx);

    // 2) recurrence: 64 WGs of 16 trajectories
    size_t smem = (size_t)BBLK * 512 * 4 + (size_t)BBLK * SHB * 2 + (size_t)BBLK * SGH * 4;
    GRU_rec_kernel<<<dim3(NTRAJ / BBLK), 256, smem, stream>>>(
        h0, dons, whh_bf, gx, bhh, out, h_last);
}
